// SQVAEQuantizer_45500883534320
// MI455X (gfx1250) — matrix operand order repaired and checked by reading, not compiled, not run on hardware
//
#include <hip/hip_runtime.h>
#include <hip/hip_bf16.h>

#define NUM_CODES 1024
#define CDIM      128
#define NROWS     65536
#define CHUNK     32      // codes per chunk (GEMM2 K)
#define NCHUNKS   (NUM_CODES / CHUNK)
#define WAVES     8
#define ZQ_ELEMS  (NROWS * CDIM)   // 8388608

typedef __attribute__((ext_vector_type(16))) __bf16  v16bf;
typedef __attribute__((ext_vector_type(8)))  float   v8f;
typedef __attribute__((ext_vector_type(4)))  unsigned int u32x4;
typedef __attribute__((ext_vector_type(8)))  int     i32x8;
typedef __attribute__((ext_vector_type(4)))  int     i32x4;

union V16 { v16bf v; unsigned short h[16]; unsigned w[8]; };
union V8F { v8f v; float f[8]; };

__device__ inline unsigned short f2bf(float f) {
  unsigned u = __builtin_bit_cast(unsigned, f);
  unsigned r = u + 0x7FFFu + ((u >> 16) & 1u);   // round-to-nearest-even
  return (unsigned short)(r >> 16);
}

// ---- TDM: DMA one 32-row x 128-col bf16 codebook chunk (8KB) into LDS ----
// Descriptor packed per CDNA5 ISA ch.8: data_size=4B, 2D tile 64 dwords x 32 rows.
__device__ inline void tdm_load_chunk(const unsigned short* Ebf, int cb, unsigned lds_off) {
  unsigned long long ga = (unsigned long long)(uintptr_t)(Ebf + (size_t)cb * CDIM);
  u32x4 g0;
  g0[0] = 1u;                                            // count=1, is_restore=0, no gather
  g0[1] = lds_off;                                       // lds_addr (bytes)
  g0[2] = (unsigned)(ga & 0xFFFFFFFFu);                  // global_addr[31:0]
  g0[3] = (unsigned)((ga >> 32) & 0x01FFFFFFu) | (2u << 30); // addr[56:32] | type=2
  i32x8 g1;
  g1[0] = (int)(2u << 16);    // workgroup_mask=0, data_size=2 (4 bytes)
  g1[1] = (int)(64u << 16);   // tensor_dim0 = 64 dwords (one 128-bf16 row)
  g1[2] = (int)(32u << 16);   // tensor_dim1 = 32 rows
  g1[3] = (int)(64u << 16);   // tile_dim0 = 64
  g1[4] = 32;                 // tile_dim1 = 32, tile_dim2 = 0
  g1[5] = 64;                 // tensor_dim0_stride = 64 dwords
  g1[6] = 0;
  g1[7] = 0;
  i32x4 zz4 = {0, 0, 0, 0};
  i32x8 zz8 = {0, 0, 0, 0, 0, 0, 0, 0};
  __builtin_amdgcn_tensor_load_to_lds(g0, g1, zz4, zz4, zz8, 0);
}

// ---------------- prep: E f32 -> bf16, code norms, zero avg_probs ----------------
__global__ __launch_bounds__(256) void sqvae_prep(const float* __restrict__ E,
                                                  unsigned short* __restrict__ Ebf,
                                                  float* __restrict__ cnorm,
                                                  float* __restrict__ avg) {
  int k = blockIdx.x * blockDim.x + threadIdx.x;
  if (k < NUM_CODES) {
    float s = 0.f;
    #pragma unroll 4
    for (int c = 0; c < CDIM; ++c) {
      float e = E[k * CDIM + c];
      s += e * e;
      Ebf[k * CDIM + c] = f2bf(e);
    }
    cnorm[k] = s;
    avg[k] = 0.f;
  }
}

// ---------------- main fused kernel ----------------
__global__ __launch_bounds__(256) void sqvae_main(const float* __restrict__ z,
                                                  const float* __restrict__ gnoise,
                                                  const float* __restrict__ temp_p,
                                                  const unsigned short* __restrict__ Ebf,
                                                  const float* __restrict__ cnorm,
                                                  float* __restrict__ avg_glob,
                                                  float* __restrict__ out) {
  __shared__ unsigned short Ebuf[2][CHUNK][CDIM];   // 16 KB double-buffered codebook chunk
  __shared__ unsigned short Pbuf[WAVES][16][CHUNK]; // 8 KB soft-weight staging (C->A relayout)
  __shared__ float avg_part[NUM_CODES];             // 4 KB per-WG avg_probs partial

  const int tid  = threadIdx.x;
  const int wave = tid >> 5;
  const int lane = tid & 31;
  const int h    = lane >> 4;      // lane half
  const int n16  = lane & 15;
  const int tile = (blockIdx.x * WAVES + wave) * 16;  // first row of this wave's tile

  for (int i = tid; i < NUM_CODES; i += 256) avg_part[i] = 0.f;

  const float inv_temp = 1.0f / temp_p[0];

  // ---- load A operand (16 rows x 128 dims of z, bf16) + row norms ----
  V16 A[4];
  float rpart = 0.f;
  {
    const int n = tile + n16;
    const long zb = (long)(n >> 12) * (CDIM * 4096) + (n & 4095);
    #pragma unroll
    for (int kc = 0; kc < 4; ++kc) {
      #pragma unroll
      for (int e = 0; e < 16; ++e) {
        int c = kc * 32 + h * 16 + e; // A-matrix 16x32 bf16, contiguous k order
        float f = z[zb + (long)c * 4096];
        rpart += f * f;
        A[kc].h[e] = f2bf(f);
      }
    }
  }
  rpart += __shfl_xor(rpart, 16);      // lane and lane^16 cover complementary K halves
  float rn[8];
  #pragma unroll
  for (int v = 0; v < 8; ++v) rn[v] = __shfl(rpart, v + h * 8);  // rnorm[M], M = v + 8h

  // bv doubles as the probs-softmax running max (argmax value == softmax max)
  float sp[8], mq[8], sq[8], bv[8];
  int   bi[8];
  #pragma unroll
  for (int v = 0; v < 8; ++v) {
    sp[v] = 0.f;
    mq[v] = -3.0e38f; sq[v] = 0.f;
    bv[v] = -3.0e38f; bi[v] = 0x7FFFFFFF;
  }

  // =========================== PASS 1: online softmax stats ===========================
  if (wave == 0) tdm_load_chunk(Ebf, 0, (unsigned)(uintptr_t)&Ebuf[0][0][0]);
  for (int j = 0; j < NCHUNKS; ++j) {
    const int cb  = j * CHUNK;
    const int buf = j & 1;
    if (wave == 0) {
      if (j + 1 < NCHUNKS) {
        tdm_load_chunk(Ebf, cb + CHUNK, (unsigned)(uintptr_t)&Ebuf[(j + 1) & 1][0][0]);
        __builtin_amdgcn_s_wait_tensorcnt(1);
      } else {
        __builtin_amdgcn_s_wait_tensorcnt(0);
      }
    }
    __syncthreads();                                  // Ebuf[buf] ready
    if (j + 1 < NCHUNKS)                              // warm L2 for next gumbel slab
      __builtin_prefetch(&gnoise[(long)(tile + n16) * NUM_CODES + cb + CHUNK], 0, 1);

    #pragma unroll
    for (int sub = 0; sub < 2; ++sub) {
      const int code = cb + sub * 16 + n16;
      const float cn = cnorm[code];
      V8F S; S.v = (v8f){};
      #pragma unroll
      for (int kc = 0; kc < 4; ++kc) {
        V16 B;
        const unsigned* src =
            (const unsigned*)&Ebuf[buf][sub * 16 + n16][kc * 32 + h * 16];
        #pragma unroll
        for (int i = 0; i < 8; ++i) B.w[i] = src[i];
        S.v = __builtin_amdgcn_wmma_f32_16x16x32_bf16(false, A[kc].v, false, B.v,
                                                      (short)0, S.v, false, false);
      }
      #pragma unroll
      for (int v = 0; v < 8; ++v) {
        float lg = 2.f * S.f[v] - rn[v] - cn;         // logit = -dist
        // branchless online softmax (probs stream); bv is the running max
        float mnew = fmaxf(bv[v], lg);
        sp[v] = sp[v] * __expf(bv[v] - mnew) + __expf(lg - mnew);
        bi[v] = (lg > bv[v]) ? code : bi[v];          // first-max wins (codes ascending)
        bv[v] = mnew;
        // branchless online softmax (gumbel stream)
        float g   = gnoise[(long)(tile + v + 8 * h) * NUM_CODES + code];
        float lq  = (lg + g) * inv_temp;
        float mqn = fmaxf(mq[v], lq);
        sq[v] = sq[v] * __expf(mq[v] - mqn) + __expf(lq - mqn);
        mq[v] = mqn;
      }
    }
    __syncthreads();                                  // done with Ebuf[buf]
  }

  // ---- cross-lane merge over the 16 lanes sharing each row ----
  #pragma unroll
  for (int v = 0; v < 8; ++v) {
    #pragma unroll
    for (int mask = 1; mask < 16; mask <<= 1) {
      float obv = __shfl_xor(bv[v], mask);
      float os  = __shfl_xor(sp[v], mask);
      int   obi = __shfl_xor(bi[v], mask);
      float M = fmaxf(bv[v], obv);
      sp[v] = sp[v] * __expf(bv[v] - M) + os * __expf(obv - M);
      bi[v] = (obv > bv[v] || (obv == bv[v] && obi < bi[v])) ? obi : bi[v];
      bv[v] = M;
      float om = __shfl_xor(mq[v], mask);
      os = __shfl_xor(sq[v], mask);
      M = fmaxf(mq[v], om);
      sq[v] = sq[v] * __expf(mq[v] - M) + os * __expf(om - M);
      mq[v] = M;
    }
  }
  if (n16 == 0) {                                     // lanes 0 & 16 write 8 rows each
    #pragma unroll
    for (int v = 0; v < 8; ++v)
      out[ZQ_ELEMS + 1 + tile + v + 8 * h] = (float)bi[v];
  }
  float isp[8], isq[8];
  #pragma unroll
  for (int v = 0; v < 8; ++v) { isp[v] = 1.f / sp[v]; isq[v] = 1.f / sq[v]; }

  V8F acc[8];
  #pragma unroll
  for (int dt = 0; dt < 8; ++dt) acc[dt].v = (v8f){};

  // ======================= PASS 2: probs, avg_probs, z_q GEMM =======================
  if (wave == 0) tdm_load_chunk(Ebf, 0, (unsigned)(uintptr_t)&Ebuf[0][0][0]);
  for (int j = 0; j < NCHUNKS; ++j) {
    const int cb  = j * CHUNK;
    const int buf = j & 1;
    if (wave == 0) {
      if (j + 1 < NCHUNKS) {
        tdm_load_chunk(Ebf, cb + CHUNK, (unsigned)(uintptr_t)&Ebuf[(j + 1) & 1][0][0]);
        __builtin_amdgcn_s_wait_tensorcnt(1);
      } else {
        __builtin_amdgcn_s_wait_tensorcnt(0);
      }
    }
    __syncthreads();                                  // Ebuf[buf] ready

    #pragma unroll
    for (int sub = 0; sub < 2; ++sub) {
      const int code = cb + sub * 16 + n16;
      const float cn = cnorm[code];
      V8F S; S.v = (v8f){};
      #pragma unroll
      for (int kc = 0; kc < 4; ++kc) {
        V16 B;
        const unsigned* src =
            (const unsigned*)&Ebuf[buf][sub * 16 + n16][kc * 32 + h * 16];
        #pragma unroll
        for (int i = 0; i < 8; ++i) B.w[i] = src[i];
        S.v = __builtin_amdgcn_wmma_f32_16x16x32_bf16(false, A[kc].v, false, B.v,
                                                      (short)0, S.v, false, false);
      }
      float psum = 0.f;
      #pragma unroll
      for (int v = 0; v < 8; ++v) {
        float lg = 2.f * S.f[v] - rn[v] - cn;
        psum += __expf(lg - bv[v]) * isp[v];          // probs for avg_probs
        float g  = gnoise[(long)(tile + v + 8 * h) * NUM_CODES + code];
        float w  = __expf((lg + g) * inv_temp - mq[v]) * isq[v]; // soft one-hot
        Pbuf[wave][v + 8 * h][sub * 16 + n16] = f2bf(w);
      }
      atomicAdd(&avg_part[code], psum);               // ds_add_f32
    }
    __syncthreads();                                  // Pbuf published

    // GEMM2: z_q[16x128] += P(16x32) @ E_chunk(32x128)
    V16 A2;
    #pragma unroll
    for (int i = 0; i < 4; ++i) {
      A2.w[i]     = *(const unsigned*)&Pbuf[wave][n16][h * 8 + 2 * i];
      A2.w[4 + i] = *(const unsigned*)&Pbuf[wave][n16][16 + h * 8 + 2 * i];
    }
    #pragma unroll
    for (int dt = 0; dt < 8; ++dt) {
      V16 B2;
      #pragma unroll
      for (int e = 0; e < 16; ++e)
        B2.h[e] = Ebuf[buf][(e & 7) + ((e >> 3) << 4) + h * 8][dt * 16 + n16];
      acc[dt].v = __builtin_amdgcn_wmma_f32_16x16x32_bf16(false, A2.v, false, B2.v,
                                                          (short)0, acc[dt].v, false, false);
    }
    __syncthreads();                                  // done with Ebuf[buf] / Pbuf
  }

  // ---- write z_q back in (B,C,H,W) layout ----
  #pragma unroll
  for (int dt = 0; dt < 8; ++dt) {
    #pragma unroll
    for (int v = 0; v < 8; ++v) {
      int n = tile + v + 8 * h;
      int c = dt * 16 + n16;
      out[(long)(n >> 12) * (CDIM * 4096) + (long)c * 4096 + (n & 4095)] = acc[dt].f[v];
    }
  }

  __syncthreads();
  for (int i = tid; i < NUM_CODES; i += 256) atomicAdd(&avg_glob[i], avg_part[i]);
}

// ---------------- perplexity reduction ----------------
__global__ __launch_bounds__(256) void sqvae_perp(const float* __restrict__ avg_glob,
                                                  float* __restrict__ out) {
  __shared__ float red[256];
  float s = 0.f;
  for (int k = threadIdx.x; k < NUM_CODES; k += 256) {
    float a = avg_glob[k] * (1.0f / (float)NROWS);
    s += a * __logf(a + 1e-10f);
  }
  red[threadIdx.x] = s;
  __syncthreads();
  for (int off = 128; off; off >>= 1) {
    if (threadIdx.x < off) red[threadIdx.x] += red[threadIdx.x + off];
    __syncthreads();
  }
  if (threadIdx.x == 0) out[ZQ_ELEMS] = __expf(-red[0]);
}

extern "C" void kernel_launch(void* const* d_in, const int* in_sizes, int n_in,
                              void* d_out, int out_size, void* d_ws, size_t ws_size,
                              hipStream_t stream) {
  const float* z     = (const float*)d_in[0];
  const float* E     = (const float*)d_in[1];
  const float* g     = (const float*)d_in[2];
  const float* temp  = (const float*)d_in[3];
  float* out = (float*)d_out;

  unsigned short* Ebf = (unsigned short*)d_ws;                       // 256 KB bf16 codebook
  float* cnorm = (float*)((char*)d_ws + NUM_CODES * CDIM * 2);       // 4 KB
  float* avg   = (float*)((char*)d_ws + NUM_CODES * CDIM * 2 + NUM_CODES * 4); // 4 KB

  sqvae_prep<<<4, 256, 0, stream>>>(E, Ebf, cnorm, avg);
  sqvae_main<<<NROWS / (WAVES * 16), 256, 0, stream>>>(z, g, temp, Ebf, cnorm, avg, out);
  sqvae_perp<<<1, 256, 0, stream>>>(avg, out);
}
